// ContrastiveAttFPLayer_40381282517154
// MI455X (gfx1250) — matrix operand order repaired and checked by reading, not compiled, not run on hardware
//
#include <hip/hip_runtime.h>
#include <hip/hip_bf16.h>
#include <math.h>

#define NN 10000
#define EE 160000
#define FF 128
#define UU 128
#define HH 8
#define BN_EPS 1e-3f

typedef __bf16 bf16_t;
typedef bf16_t v16bf __attribute__((ext_vector_type(16)));
typedef bf16_t v8bf  __attribute__((ext_vector_type(8)));
typedef float  v8f   __attribute__((ext_vector_type(8)));

// ---------------- WMMA fragment helpers (CDNA5 16x16x32 bf16 layouts) ----------------
// A (16x32, MxK): lane l -> row M=l%16, half=l/16.
//   elements 0..15 : K = half*16 + e         (contiguous 32B)
__device__ __forceinline__ v16bf load_a_frag(const bf16_t* rowPtr, int half) {
    return *reinterpret_cast<const v16bf*>(rowPtr + half * 16);
}
// B (32x16, KxN): lane l -> col N=l%16, K = e + (l/16)*16. Pre-packed contiguous 32B per lane.
__device__ __forceinline__ v16bf load_b_frag(const bf16_t* packBase, int kc, int ntilesTot,
                                             int nt, int lane) {
    const bf16_t* p = packBase + ((size_t)(kc * ntilesTot + nt) * 32 + lane) * 16;
    return *reinterpret_cast<const v16bf*>(p);
}
__device__ __forceinline__ v8f wmma_bf16(v16bf a, v16bf b, v8f c) {
    return __builtin_amdgcn_wmma_f32_16x16x32_bf16(false, a, false, b, (short)0, c, false, false);
}
// order-preserving float->u32 encoding for atomic segment-max
__device__ __forceinline__ unsigned encodeF(float f) {
    unsigned u = __float_as_uint(f);
    return (u & 0x80000000u) ? ~u : (u | 0x80000000u);
}
__device__ __forceinline__ float decodeF(unsigned u) {
    unsigned b = (u & 0x80000000u) ? (u & 0x7FFFFFFFu) : ~u;
    return __uint_as_float(b);
}

// ---------------- prep kernels ----------------
__global__ void cvt_bf16_kernel(const float* __restrict__ src, bf16_t* __restrict__ dst, int n) {
    int i = blockIdx.x * blockDim.x + threadIdx.x;
    if (i < n) dst[i] = (bf16_t)src[i];
}

// Pack row-major W (Ktot x Ntot) into per-lane WMMA B-fragment order.
// attMode: W is W_att (H,F,U); logical col c -> (h=c/U, u=c%U), element W[h, K, u].
__global__ void pack_b_kernel(const float* __restrict__ W, bf16_t* __restrict__ dst,
                              int kchunks, int ntiles, int Ntot, int attMode) {
    int idx = blockIdx.x * blockDim.x + threadIdx.x;
    int total = kchunks * ntiles * 32 * 16;
    if (idx >= total) return;
    int e  = idx & 15;
    int l  = (idx >> 4) & 31;
    int nt = (idx >> 9) % ntiles;
    int kc = (idx >> 9) / ntiles;
    int K    = kc * 32 + ((l >> 4) << 4) + e;
    int Ncol = nt * 16 + (l & 15);
    float v;
    if (attMode) {
        int h = Ncol / UU, u = Ncol % UU;
        v = W[((size_t)h * FF + K) * UU + u];
    } else {
        v = W[(size_t)K * Ntot + Ncol];
    }
    dst[idx] = (bf16_t)v;
}

// ---------------- GEMM kernels (1 wave = 16 rows x 128 cols tile; 96 cols in gru_gemm) ----------------
// edge_updated_pre = relu([node[src] | node[dst] | edge_attr] @ W_edge + b_edge), + column stats
__global__ void edge_gemm_kernel(const bf16_t* __restrict__ nodeBF, const bf16_t* __restrict__ edgeBF,
                                 const int* __restrict__ eidx, const bf16_t* __restrict__ WedgePK,
                                 const float* __restrict__ b_edge, float* __restrict__ edgePre,
                                 float* __restrict__ colSum, float* __restrict__ colSumSq) {
    int lane = threadIdx.x, half = lane >> 4, r = lane & 15;
    int tile = blockIdx.x;
    int e = tile * 16 + r;
    int di = eidx[2 * e + 0];
    int si = eidx[2 * e + 1];
    const bf16_t* rp0 = nodeBF + (size_t)si * FF;   // cols   0..127 : node[src]
    const bf16_t* rp1 = nodeBF + (size_t)di * FF;   // cols 128..255 : node[dst]
    const bf16_t* rp2 = edgeBF + (size_t)e  * FF;   // cols 256..383 : edge_attr
    v8f acc[8] = {};
#pragma unroll
    for (int kc = 0; kc < 12; ++kc) {
        const bf16_t* rowPtr = (kc < 4 ? rp0 : (kc < 8 ? rp1 : rp2)) + (kc & 3) * 32;
        v16bf a = load_a_frag(rowPtr, half);
        v16bf bfr[8];
#pragma unroll
        for (int nt = 0; nt < 8; ++nt) bfr[nt] = load_b_frag(WedgePK, kc, 8, nt, lane);
#pragma unroll
        for (int nt = 0; nt < 8; ++nt) acc[nt] = wmma_bf16(a, bfr[nt], acc[nt]);
    }
#pragma unroll
    for (int nt = 0; nt < 8; ++nt) {
        int col = nt * 16 + r;
        float bb = b_edge[col];
        float s = 0.f, sq = 0.f;
#pragma unroll
        for (int rr = 0; rr < 8; ++rr) {
            int M = rr + half * 8;
            float v = acc[nt][rr] + bb;
            v = v > 0.f ? v : 0.f;
            edgePre[(size_t)(tile * 16 + M) * UU + col] = v;
            s += v; sq += v * v;
        }
        s  += __shfl_xor(s, 16);
        sq += __shfl_xor(sq, 16);
        if (half == 0) { atomicAdd(&colSum[col], s); atomicAdd(&colSumSq[col], sq); }
    }
}

// z[h,n,u] = node @ W_att[h] + b_att[h]
__global__ void z_gemm_kernel(const bf16_t* __restrict__ nodeBF, const bf16_t* __restrict__ WcatPK,
                              const float* __restrict__ b_att, float* __restrict__ zbuf) {
    int lane = threadIdx.x, half = lane >> 4, r = lane & 15;
    int tile = blockIdx.x, h = blockIdx.y;
    const bf16_t* rp = nodeBF + (size_t)(tile * 16 + r) * FF;
    v8f acc[8] = {};
#pragma unroll
    for (int kc = 0; kc < 4; ++kc) {
        v16bf a = load_a_frag(rp + kc * 32, half);
        v16bf bfr[8];
#pragma unroll
        for (int nt = 0; nt < 8; ++nt) bfr[nt] = load_b_frag(WcatPK, kc, 64, h * 8 + nt, lane);
#pragma unroll
        for (int nt = 0; nt < 8; ++nt) acc[nt] = wmma_bf16(a, bfr[nt], acc[nt]);
    }
#pragma unroll
    for (int nt = 0; nt < 8; ++nt) {
        int col = nt * 16 + r;
        float bb = b_att[h * UU + col];
#pragma unroll
        for (int rr = 0; rr < 8; ++rr) {
            int row = tile * 16 + rr + half * 8;
            zbuf[((size_t)h * NN + row) * UU + col] = acc[nt][rr] + bb;
        }
    }
}

// node_updated_pre = relu(node @ W_node + b_node), + column stats
__global__ void node_gemm_kernel(const bf16_t* __restrict__ nodeBF, const bf16_t* __restrict__ WnodePK,
                                 const float* __restrict__ b_node, float* __restrict__ nodePre,
                                 float* __restrict__ colSum, float* __restrict__ colSumSq) {
    int lane = threadIdx.x, half = lane >> 4, r = lane & 15;
    int tile = blockIdx.x;
    const bf16_t* rp = nodeBF + (size_t)(tile * 16 + r) * FF;
    v8f acc[8] = {};
#pragma unroll
    for (int kc = 0; kc < 4; ++kc) {
        v16bf a = load_a_frag(rp + kc * 32, half);
        v16bf bfr[8];
#pragma unroll
        for (int nt = 0; nt < 8; ++nt) bfr[nt] = load_b_frag(WnodePK, kc, 8, nt, lane);
#pragma unroll
        for (int nt = 0; nt < 8; ++nt) acc[nt] = wmma_bf16(a, bfr[nt], acc[nt]);
    }
#pragma unroll
    for (int nt = 0; nt < 8; ++nt) {
        int col = nt * 16 + r;
        float bb = b_node[col];
        float s = 0.f, sq = 0.f;
#pragma unroll
        for (int rr = 0; rr < 8; ++rr) {
            int M = rr + half * 8;
            float v = acc[nt][rr] + bb;
            v = v > 0.f ? v : 0.f;
            nodePre[(size_t)(tile * 16 + M) * UU + col] = v;
            s += v; sq += v * v;
        }
        s  += __shfl_xor(s, 16);
        sq += __shfl_xor(sq, 16);
        if (half == 0) { atomicAdd(&colSum[col], s); atomicAdd(&colSumSq[col], sq); }
    }
}

// mx = attended @ gru_kernel + bias0 ; mi = updated @ gru_rkernel + bias1
__global__ void gru_gemm_kernel(const bf16_t* __restrict__ attBF, const bf16_t* __restrict__ updBF,
                                const bf16_t* __restrict__ GxPK, const bf16_t* __restrict__ GhPK,
                                const float* __restrict__ gru_bias,
                                float* __restrict__ mx, float* __restrict__ mi) {
    int lane = threadIdx.x, half = lane >> 4, r = lane & 15;
    int tile = blockIdx.x, grp = blockIdx.y, which = blockIdx.z;
    const bf16_t* A   = which ? updBF : attBF;
    const bf16_t* BPK = which ? GhPK  : GxPK;
    const float* bias = gru_bias + which * 384;
    float* out        = which ? mi : mx;
    const bf16_t* rp = A + (size_t)(tile * 16 + r) * UU;
    v8f acc[6] = {};
#pragma unroll
    for (int kc = 0; kc < 4; ++kc) {
        v16bf a = load_a_frag(rp + kc * 32, half);
        v16bf bfr[6];
#pragma unroll
        for (int nt = 0; nt < 6; ++nt) bfr[nt] = load_b_frag(BPK, kc, 24, grp * 6 + nt, lane);
#pragma unroll
        for (int nt = 0; nt < 6; ++nt) acc[nt] = wmma_bf16(a, bfr[nt], acc[nt]);
    }
#pragma unroll
    for (int nt = 0; nt < 6; ++nt) {
        int col = (grp * 6 + nt) * 16 + r;
        float bb = bias[col];
#pragma unroll
        for (int rr = 0; rr < 8; ++rr) {
            int row = tile * 16 + rr + half * 8;
            out[(size_t)row * 384 + col] = acc[nt][rr] + bb;
        }
    }
}

// ---------------- elementwise / reduction kernels ----------------
// stats layout (floats): [0:128)=eSum [128:256)=eSq [256:384)=nSum [384:512)=nSq
// scale/shift (at stats+512): [0:128)=scaleE [128)=shiftE [256)=scaleN [384)=shiftN
__global__ void bn_finalize_kernel(const float* __restrict__ stats,
                                   const float* __restrict__ gamma_e, const float* __restrict__ beta_e,
                                   const float* __restrict__ gamma_n, const float* __restrict__ beta_n,
                                   float* __restrict__ ss) {
    int c = threadIdx.x;
    if (blockIdx.x == 0) {
        float mean = stats[c] / (float)EE;
        float var  = stats[128 + c] / (float)EE - mean * mean;
        float sc = gamma_e[c] * rsqrtf(var + BN_EPS);
        ss[c] = sc; ss[128 + c] = beta_e[c] - mean * sc;
    } else {
        float mean = stats[256 + c] / (float)NN;
        float var  = stats[384 + c] / (float)NN - mean * mean;
        float sc = gamma_n[c] * rsqrtf(var + BN_EPS);
        ss[256 + c] = sc; ss[384 + c] = beta_n[c] - mean * sc;
    }
}

// BN-apply edge rows -> d_out, plus s_edge[h,e] = edge_updated . a_edge[h]
__global__ void edge_norm_score_kernel(const float* __restrict__ edgePre, const float* __restrict__ ss,
                                       const float* __restrict__ a_att, float* __restrict__ outEdge,
                                       float* __restrict__ sEdge) {
    int e = blockIdx.x, c = threadIdx.x;
    float v = edgePre[(size_t)e * UU + c] * ss[c] + ss[128 + c];
    outEdge[(size_t)e * UU + c] = v;
    float p[HH];
#pragma unroll
    for (int h = 0; h < HH; ++h) p[h] = v * a_att[h * 384 + 256 + c];
    __shared__ float sred[HH][4];
    int lane = c & 31, wv = c >> 5;
#pragma unroll
    for (int h = 0; h < HH; ++h) {
        float t = p[h];
        for (int o = 16; o > 0; o >>= 1) t += __shfl_xor(t, o);
        if (lane == 0) sred[h][wv] = t;
    }
    __syncthreads();
    if (c < HH) sEdge[(size_t)c * EE + e] = sred[c][0] + sred[c][1] + sred[c][2] + sred[c][3];
}

// sA[h,n] = z[h,n,:] . a_src[h] ; sD[h,n] = z[h,n,:] . a_dst[h]
__global__ void node_score_kernel(const float* __restrict__ zbuf, const float* __restrict__ a_att,
                                  float* __restrict__ sA, float* __restrict__ sD) {
    int n = blockIdx.x, c = threadIdx.x;
    float pa[HH], pd[HH];
#pragma unroll
    for (int h = 0; h < HH; ++h) {
        float zv = zbuf[((size_t)h * NN + n) * UU + c];
        pa[h] = zv * a_att[h * 384 + c];
        pd[h] = zv * a_att[h * 384 + 128 + c];
    }
    __shared__ float ra[HH][4], rd[HH][4];
    int lane = c & 31, wv = c >> 5;
#pragma unroll
    for (int h = 0; h < HH; ++h) {
        float ta = pa[h], td = pd[h];
        for (int o = 16; o > 0; o >>= 1) { ta += __shfl_xor(ta, o); td += __shfl_xor(td, o); }
        if (lane == 0) { ra[h][wv] = ta; rd[h][wv] = td; }
    }
    __syncthreads();
    if (c < HH)            sA[(size_t)c * NN + n]        = ra[c][0] + ra[c][1] + ra[c][2] + ra[c][3];
    else if (c < 2 * HH)   sD[(size_t)(c - HH) * NN + n] = rd[c - HH][0] + rd[c - HH][1] + rd[c - HH][2] + rd[c - HH][3];
}

__global__ void logit_max_kernel(const int* __restrict__ eidx, const float* __restrict__ sA,
                                 const float* __restrict__ sD, const float* __restrict__ sEdge,
                                 float* __restrict__ logitBuf, unsigned* __restrict__ segMax) {
    int e = blockIdx.x * blockDim.x + threadIdx.x;
    int h = blockIdx.y;
    if (e >= EE) return;
    int di = eidx[2 * e], si = eidx[2 * e + 1];
    float s = sA[(size_t)h * NN + si] + sD[(size_t)h * NN + di] + sEdge[(size_t)h * EE + e];
    float l = s > 0.f ? s : 0.2f * s;
    logitBuf[(size_t)h * EE + e] = l;
    atomicMax(&segMax[(size_t)h * NN + di], encodeF(l));
}

__global__ void exp_den_kernel(const int* __restrict__ eidx, const unsigned* __restrict__ segMax,
                               float* __restrict__ logitBuf, float* __restrict__ den) {
    int e = blockIdx.x * blockDim.x + threadIdx.x;
    int h = blockIdx.y;
    if (e >= EE) return;
    int di = eidx[2 * e];
    float m = decodeF(segMax[(size_t)h * NN + di]);
    float v = expf(logitBuf[(size_t)h * EE + e] - m);
    logitBuf[(size_t)h * EE + e] = v;
    atomicAdd(&den[(size_t)h * NN + di], v);
}

__global__ void aggregate_kernel(const int* __restrict__ eidx, const float* __restrict__ exBuf,
                                 const float* __restrict__ den, const float* __restrict__ zbuf,
                                 float* __restrict__ agg) {
    int e = blockIdx.x, h = blockIdx.y, c = threadIdx.x;
    int di = eidx[2 * e], si = eidx[2 * e + 1];
    float alpha = exBuf[(size_t)h * EE + e] / den[(size_t)h * NN + di];
    atomicAdd(&agg[((size_t)h * NN + di) * UU + c], alpha * zbuf[((size_t)h * NN + si) * UU + c]);
}

__global__ void attended_upd_kernel(const float* __restrict__ agg, const float* __restrict__ nodePre,
                                    const float* __restrict__ ss, bf16_t* __restrict__ attBF,
                                    bf16_t* __restrict__ updBF, float* __restrict__ updF) {
    int n = blockIdx.x, c = threadIdx.x;
    float a = 0.f;
#pragma unroll
    for (int h = 0; h < HH; ++h) a += agg[((size_t)h * NN + n) * UU + c];
    a *= (1.0f / HH);
    a = a > 0.f ? a : 0.f;
    attBF[(size_t)n * UU + c] = (bf16_t)a;
    float u = nodePre[(size_t)n * UU + c] * ss[256 + c] + ss[384 + c];
    updF[(size_t)n * UU + c] = u;
    updBF[(size_t)n * UU + c] = (bf16_t)u;
}

__global__ void gru_combine_kernel(const float* __restrict__ mx, const float* __restrict__ mi,
                                   const float* __restrict__ updF, float* __restrict__ outNode) {
    int n = blockIdx.x, c = threadIdx.x;
    size_t b = (size_t)n * 384;
    float xz = mx[b + c], xr = mx[b + 128 + c], xh = mx[b + 256 + c];
    float rz = mi[b + c], rr = mi[b + 128 + c], rh = mi[b + 256 + c];
    float zg = 1.f / (1.f + expf(-(xz + rz)));
    float rg = 1.f / (1.f + expf(-(xr + rr)));
    float hh = tanhf(xh + rg * rh);
    float u = updF[(size_t)n * UU + c];
    outNode[(size_t)n * UU + c] = zg * u + (1.f - zg) * hh;
}

// ---------------- host launcher ----------------
extern "C" void kernel_launch(void* const* d_in, const int* in_sizes, int n_in,
                              void* d_out, int out_size, void* d_ws, size_t ws_size,
                              hipStream_t stream) {
    (void)in_sizes; (void)n_in; (void)out_size; (void)ws_size;
    const float* node_attr = (const float*)d_in[0];
    const float* edge_attr = (const float*)d_in[1];
    const int*   eidx      = (const int*)d_in[2];
    const float* W_edge  = (const float*)d_in[3];
    const float* b_edge  = (const float*)d_in[4];
    const float* gamma_e = (const float*)d_in[5];
    const float* beta_e  = (const float*)d_in[6];
    const float* W_node  = (const float*)d_in[7];
    const float* b_node  = (const float*)d_in[8];
    const float* gamma_n = (const float*)d_in[9];
    const float* beta_n  = (const float*)d_in[10];
    const float* W_att   = (const float*)d_in[11];
    const float* b_att   = (const float*)d_in[12];
    const float* a_att   = (const float*)d_in[13];
    const float* gru_k   = (const float*)d_in[14];
    const float* gru_rk  = (const float*)d_in[15];
    const float* gru_b   = (const float*)d_in[16];

    char* ws = (char*)d_ws;
    size_t off = 0;
    auto alloc = [&](size_t bytes) -> char* {
        char* p = ws + off;
        off += (bytes + 255) & ~(size_t)255;
        return p;
    };
    bf16_t* nodeBF  = (bf16_t*)alloc((size_t)NN * FF * 2);
    bf16_t* edgeBF  = (bf16_t*)alloc((size_t)EE * FF * 2);
    bf16_t* WedgePK = (bf16_t*)alloc((size_t)384 * 128 * 2);
    bf16_t* WcatPK  = (bf16_t*)alloc((size_t)128 * 1024 * 2);
    bf16_t* WnodePK = (bf16_t*)alloc((size_t)128 * 128 * 2);
    bf16_t* GxPK    = (bf16_t*)alloc((size_t)128 * 384 * 2);
    bf16_t* GhPK    = (bf16_t*)alloc((size_t)128 * 384 * 2);
    float*  statsF  = (float*)alloc(1024 * 4);                 // 512 sums + 512 scale/shift
    float*  zbuf    = (float*)alloc((size_t)HH * NN * UU * 4);
    float*  sEdge   = (float*)alloc((size_t)HH * EE * 4);
    float*  sA      = (float*)alloc((size_t)HH * NN * 4);
    float*  sD      = (float*)alloc((size_t)HH * NN * 4);
    float*  exBuf   = (float*)alloc((size_t)HH * EE * 4);
    unsigned* segMax = (unsigned*)alloc((size_t)HH * NN * 4);
    float*  den     = (float*)alloc((size_t)HH * NN * 4);
    float*  nodePre = (float*)alloc((size_t)NN * UU * 4);
    float*  agg     = (float*)alloc((size_t)HH * NN * UU * 4);
    char*   edgePreBase = alloc((size_t)EE * UU * 4);
    float*  edgePre = (float*)edgePreBase;
    // After edge_norm_score consumes edgePre, reuse its 82MB for GRU-stage buffers (40.96MB):
    float*  mx    = (float*)edgePreBase;
    float*  mi    = (float*)(edgePreBase + (size_t)NN * 384 * 4);
    bf16_t* attBF = (bf16_t*)(edgePreBase + 2 * (size_t)NN * 384 * 4);
    bf16_t* updBF = (bf16_t*)(edgePreBase + 2 * (size_t)NN * 384 * 4 + (size_t)NN * UU * 2);
    float*  updF  = (float*)(edgePreBase + 2 * (size_t)NN * 384 * 4 + 2 * (size_t)NN * UU * 2);

    // per-call zeroing of accumulators (graph-capture safe)
    hipMemsetAsync(statsF, 0, 512 * 4, stream);
    hipMemsetAsync(segMax, 0, (size_t)HH * NN * 4, stream);
    hipMemsetAsync(den,    0, (size_t)HH * NN * 4, stream);
    hipMemsetAsync(agg,    0, (size_t)HH * NN * UU * 4, stream);

    const int T = 256;
    cvt_bf16_kernel<<<(NN * FF + T - 1) / T, T, 0, stream>>>(node_attr, nodeBF, NN * FF);
    cvt_bf16_kernel<<<(EE * FF + T - 1) / T, T, 0, stream>>>(edge_attr, edgeBF, EE * FF);

    pack_b_kernel<<<(12 * 8 * 512 + T - 1) / T, T, 0, stream>>>(W_edge, WedgePK, 12, 8, 128, 0);
    pack_b_kernel<<<(4 * 64 * 512 + T - 1) / T, T, 0, stream>>>(W_att, WcatPK, 4, 64, 1024, 1);
    pack_b_kernel<<<(4 * 8 * 512 + T - 1) / T, T, 0, stream>>>(W_node, WnodePK, 4, 8, 128, 0);
    pack_b_kernel<<<(4 * 24 * 512 + T - 1) / T, T, 0, stream>>>(gru_k, GxPK, 4, 24, 384, 0);
    pack_b_kernel<<<(4 * 24 * 512 + T - 1) / T, T, 0, stream>>>(gru_rk, GhPK, 4, 24, 384, 0);

    edge_gemm_kernel<<<EE / 16, 32, 0, stream>>>(nodeBF, edgeBF, eidx, WedgePK, b_edge,
                                                 edgePre, statsF + 0, statsF + 128);
    z_gemm_kernel<<<dim3(NN / 16, HH), 32, 0, stream>>>(nodeBF, WcatPK, b_att, zbuf);
    node_gemm_kernel<<<NN / 16, 32, 0, stream>>>(nodeBF, WnodePK, b_node, nodePre,
                                                 statsF + 256, statsF + 384);
    bn_finalize_kernel<<<2, 128, 0, stream>>>(statsF, gamma_e, beta_e, gamma_n, beta_n, statsF + 512);

    float* outNode = (float*)d_out;
    float* outEdge = outNode + (size_t)NN * UU;
    edge_norm_score_kernel<<<EE, 128, 0, stream>>>(edgePre, statsF + 512, a_att, outEdge, sEdge);
    node_score_kernel<<<NN, 128, 0, stream>>>(zbuf, a_att, sA, sD);
    logit_max_kernel<<<dim3((EE + 255) / 256, HH), 256, 0, stream>>>(eidx, sA, sD, sEdge, exBuf, segMax);
    exp_den_kernel<<<dim3((EE + 255) / 256, HH), 256, 0, stream>>>(eidx, segMax, exBuf, den);
    aggregate_kernel<<<dim3(EE, HH), 128, 0, stream>>>(eidx, exBuf, den, zbuf, agg);
    attended_upd_kernel<<<NN, 128, 0, stream>>>(agg, nodePre, statsF + 512, attBF, updBF, updF);
    gru_gemm_kernel<<<dim3(NN / 16, 4, 2), 32, 0, stream>>>(attBF, updBF, GxPK, GhPK, gru_b, mx, mi);
    gru_combine_kernel<<<NN, 128, 0, stream>>>(mx, mi, updF, outNode);
}
